// ModelForQuestionAnsweringV2_65833258713775
// MI455X (gfx1250) — matrix operand order repaired and checked by reading, not compiled, not run on hardware
//
#include <hip/hip_runtime.h>
#include <hip/hip_bf16.h>
#include <math.h>

// ---------------------------------------------------------------------------
// Types for CDNA5 WMMA / TDM
// ---------------------------------------------------------------------------
typedef unsigned int u32;
typedef unsigned short u16;
typedef __attribute__((ext_vector_type(16))) __bf16 v16bf;
typedef __attribute__((ext_vector_type(8)))  float  v8f;
typedef __attribute__((ext_vector_type(4))) unsigned int u32x4;
typedef __attribute__((ext_vector_type(8))) int i32x8;
typedef __attribute__((ext_vector_type(4))) int i32x4;

union FragBF16 { v16bf v; u32 u[8]; };
union AccF32   { v8f v; float f[8]; };

__device__ __forceinline__ u16 f2bf(float x) {
    union { float f; u32 u; } v; v.f = x;
    u32 u = v.u;
    return (u16)((u + 0x7FFFu + ((u >> 16) & 1u)) >> 16);  // RNE
}
__device__ __forceinline__ float bf2f(u16 h) {
    union { float f; u32 u; } v; v.u = ((u32)h) << 16; return v.f;
}
__device__ __forceinline__ float gelu_f(float x) {
    const float k = 0.7978845608028654f;
    return 0.5f * x * (1.0f + tanhf(k * (x + 0.044715f * x * x * x)));
}

// ---------------------------------------------------------------------------
// GEMM tiling parameters
// ---------------------------------------------------------------------------
#define BM 128
#define BN 128
#define BK 32
#define LDP 34   // LDS row pitch in bf16: 68B = 17 DWORDs -> conflict-free, TDM pad

// ---------------------------------------------------------------------------
// Tensor Data Mover: load one BK x BM (32 x 128) bf16 tile into LDS with
// 1-DWORD pad every 16 DWORDs (row pitch 34 bf16).  D# per CDNA5 ISA ch.8.
// ---------------------------------------------------------------------------
__device__ __forceinline__ void tdm_load_tile_bf16(u32 lds_byte_addr,
                                                   const u16* gptr,
                                                   u32 tensor_d0, u32 tensor_d1,
                                                   u32 stride0_elems)
{
    unsigned long long ga = (unsigned long long)(uintptr_t)gptr;
    u32x4 g0;
    g0[0] = 1u;                                          // count=1 (valid, user mode)
    g0[1] = lds_byte_addr;                               // LDS dest (bytes)
    g0[2] = (u32)ga;                                     // global addr lo
    g0[3] = ((u32)(ga >> 32) & 0x01FFFFFFu) | (2u << 30);// addr hi [56:32] | type=2
    i32x8 g1;
    g1[0] = (int)((1u << 16)      // data_size = 1 -> 2 bytes (bf16)
                | (1u << 20)      // pad_enable
                | (3u << 22)      // pad_interval code 3 -> every 16 DWORDs (64B row)
                | (0u << 25));    // pad_amount code 0 -> 1 DWORD
    g1[1] = (int)((tensor_d0 & 0xFFFFu) << 16);                         // dim0[15:0]
    g1[2] = (int)((tensor_d0 >> 16) | ((tensor_d1 & 0xFFFFu) << 16));   // dim0 hi, dim1 lo
    g1[3] = (int)((tensor_d1 >> 16) | ((u32)BK << 16));                 // dim1 hi, tile_dim0=32
    g1[4] = (int)BM;                                                    // tile_dim1=128, tile_dim2=0
    g1[5] = (int)stride0_elems;                                         // dim0_stride lo
    g1[6] = 0;                                                          // dim0_stride hi, dim1_stride lo
    g1[7] = 0;                                                          // dim1_stride hi
    i32x4 gz = (i32x4)0;
#if defined(__clang_major__) && (__clang_major__ >= 23)
    __builtin_amdgcn_tensor_load_to_lds(g0, g1, gz, gz, (i32x8)0, 0);
#else
    __builtin_amdgcn_tensor_load_to_lds(g0, g1, gz, gz, 0);
#endif
}

// ---------------------------------------------------------------------------
// Elementwise utility kernels
// ---------------------------------------------------------------------------
__global__ void k_zero_f32(float* __restrict__ p, int n) {
    int i = blockIdx.x * blockDim.x + threadIdx.x;
    int st = gridDim.x * blockDim.x;
    for (; i < n; i += st) p[i] = 0.0f;
}

__global__ void k_f32_to_bf16(const float* __restrict__ src, u16* __restrict__ dst, int n) {
    int i = blockIdx.x * blockDim.x + threadIdx.x;
    int st = gridDim.x * blockDim.x;
    for (; i < n; i += st) dst[i] = f2bf(src[i]);
}

// Transposed convert: src (K x N) f32 row-major -> dst (N x K) bf16 row-major
__global__ __launch_bounds__(256)
void k_f32_to_bf16_T(const float* __restrict__ src, u16* __restrict__ dst,
                     int Kd, int Nd)
{
    __shared__ float tile[32][33];
    int kb = blockIdx.x * 32, nb = blockIdx.y * 32;
    int tx = threadIdx.x & 31, ty = threadIdx.x >> 5;   // 32 x 8
    #pragma unroll
    for (int i = ty; i < 32; i += 8)
        tile[i][tx] = src[(size_t)(kb + i) * Nd + nb + tx];
    __syncthreads();
    #pragma unroll
    for (int i = ty; i < 32; i += 8)
        dst[(size_t)(nb + i) * Kd + kb + tx] = f2bf(tile[tx][i]);
}

// ---------------------------------------------------------------------------
// Segment scatter-add: col_cell / row_cell   (one block per (b,l) token)
// ---------------------------------------------------------------------------
__global__ __launch_bounds__(256)
void k_scatter_cells(const int* __restrict__ input_ids,
                     const int* __restrict__ tti,          // (B,L,3)
                     const float* __restrict__ seq,        // (B,L,768)
                     float* __restrict__ col_cell,         // (B,256,768)
                     float* __restrict__ row_cell)         // (B,256,768)
{
    int bl = blockIdx.x;                 // b*1024 + l
    int b  = bl >> 10;
    int iid = input_ids[bl];
    int cid = tti[bl * 3 + 1];
    int rid = tti[bl * 3 + 2];
    bool colm = (iid == 2) && (rid == 0);
    bool tokm = (iid > 2);
    if (!colm && !tokm) return;
    const float* s = seq + (size_t)bl * 768;
    float* cdst = col_cell + ((size_t)(b * 256 + cid)) * 768;
    float* rdst = row_cell + ((size_t)(b * 256 + rid)) * 768;
    for (int d = threadIdx.x; d < 768; d += 256) {
        float v = s[d];
        if (colm) atomicAdd(&cdst[d], v);
        if (tokm) atomicAdd(&rdst[d], v);
    }
}

// ---------------------------------------------------------------------------
// BF16 WMMA GEMM with TDM-fed, double-buffered LDS tiles.
//   A : MxK bf16 row-major          (streamed, TDM tile 32x128 rows=M)
//   WT: NxK bf16 row-major (= W^T)  (TDM tile 32x128 rows=N, lands as Bs[n][k])
//   out = [gelu](A @ W + bias)  -> MxN bf16
//   Block tile 128x128, BK=32, 8 waves (4 M x 2 N), wave tile 32x64.
// ---------------------------------------------------------------------------
__global__ __launch_bounds__(256)
void k_gemm_bf16(const u16* __restrict__ A, const u16* __restrict__ WT,
                 const float* __restrict__ bias, u16* __restrict__ out,
                 int M, int N, int K, int do_gelu)
{
    __shared__ __align__(16) u16 As[2][BM][LDP];
    __shared__ __align__(16) u16 Bs[2][BN][LDP];

    const int tid  = threadIdx.x;
    const int lane = tid & 31;
    const int wave = tid >> 5;
    const int half = lane >> 4;
    const int lr   = lane & 15;
    const int wm   = wave >> 1;   // 0..3
    const int wn   = wave & 1;    // 0..1
    const int m0   = blockIdx.y * BM;
    const int n0   = blockIdx.x * BN;

    AccF32 acc[2][4];
    #pragma unroll
    for (int i = 0; i < 2; i++)
        #pragma unroll
        for (int j = 0; j < 4; j++)
            #pragma unroll
            for (int e = 0; e < 8; e++) acc[i][j].f[e] = 0.0f;

    const int nsteps = K / BK;

    // prologue: prefetch tile pair 0 into buffer 0
    if (wave == 0) {
        tdm_load_tile_bf16((u32)(uintptr_t)&As[0][0][0], A  + (size_t)m0 * K, K, M, K);
        tdm_load_tile_bf16((u32)(uintptr_t)&Bs[0][0][0], WT + (size_t)n0 * K, K, N, K);
    }

    for (int s = 0; s < nsteps; ++s) {
        const int cur = s & 1;
        if (wave == 0) {
            if (s + 1 < nsteps) {
                int k1 = (s + 1) * BK;
                tdm_load_tile_bf16((u32)(uintptr_t)&As[cur ^ 1][0][0],
                                   A  + (size_t)m0 * K + k1, K, M, K);
                tdm_load_tile_bf16((u32)(uintptr_t)&Bs[cur ^ 1][0][0],
                                   WT + (size_t)n0 * K + k1, K, N, K);
                __builtin_amdgcn_s_wait_tensorcnt(2);   // pair s complete, pair s+1 in flight
            } else {
                __builtin_amdgcn_s_wait_tensorcnt(0);
            }
        }
        __syncthreads();   // tile pair s visible to all waves

        // ---- fragments per documented CDNA5 bf16 WMMA layouts ----
        FragBF16 afrag[2], bfrag[4];
        #pragma unroll
        for (int mi = 0; mi < 2; mi++) {
            int row = wm * 32 + mi * 16 + lr;
            #pragma unroll
            for (int v = 0; v < 8; v++) {
                // A 16x32: VGPR v slot j -> K = 16*(v/4) + 8*half + 2*(v%4) + j
                int kb = 16 * (v >> 2) + 8 * half + 2 * (v & 3);
                afrag[mi].u[v] = *(const u32*)&As[cur][row][kb];
            }
        }
        #pragma unroll
        for (int ni = 0; ni < 4; ni++) {
            int col = wn * 64 + ni * 16 + lr;
            #pragma unroll
            for (int v = 0; v < 8; v++) {
                // B 32x16: VGPR v slot j -> K = 16*(v/4) + 8*half + 2*(v%4) + j, N = lane%16
                int kb = 16 * (v >> 2) + 8 * half + 2 * (v & 3);
                bfrag[ni].u[v] = *(const u32*)&Bs[cur][col][kb];
            }
        }
        #pragma unroll
        for (int mi = 0; mi < 2; mi++)
            #pragma unroll
            for (int ni = 0; ni < 4; ni++)
                acc[mi][ni].v = __builtin_amdgcn_wmma_f32_16x16x32_bf16(
                    false, afrag[mi].v, false, bfrag[ni].v,
                    (short)0, acc[mi][ni].v, false, false);

        __syncthreads();   // all reads of buffer `cur` done before TDM overwrites it
    }

    // ---- epilogue: bias + gelu + bf16 store ----
    #pragma unroll
    for (int mi = 0; mi < 2; mi++) {
        #pragma unroll
        for (int ni = 0; ni < 4; ni++) {
            int col = n0 + wn * 64 + ni * 16 + lr;
            float bv = bias ? bias[col] : 0.0f;
            #pragma unroll
            for (int v = 0; v < 8; v++) {
                int row = m0 + wm * 32 + mi * 16 + v + 8 * half;
                float x = acc[mi][ni].f[v] + bv;
                if (do_gelu) x = gelu_f(x);
                out[(size_t)row * N + col] = f2bf(x);
            }
        }
    }
}

// ---------------------------------------------------------------------------
// prediction_column: dot(col_h[row,:], W_cls_col) + b   (one block per row)
// ---------------------------------------------------------------------------
__global__ __launch_bounds__(256)
void k_rowdot(const u16* __restrict__ h, const float* __restrict__ w,
              const float* __restrict__ b, float* __restrict__ out, int K)
{
    __shared__ float sm[256];
    int row = blockIdx.x;
    float acc = 0.0f;
    for (int k = threadIdx.x; k < K; k += 256)
        acc += bf2f(h[(size_t)row * K + k]) * w[k];
    sm[threadIdx.x] = acc;
    __syncthreads();
    for (int s = 128; s > 0; s >>= 1) {
        if (threadIdx.x < s) sm[threadIdx.x] += sm[threadIdx.x + s];
        __syncthreads();
    }
    if (threadIdx.x == 0) out[row] = sm[0] + b[0];
}

// ---------------------------------------------------------------------------
// Build h = concat(col_hidden, row_hidden, seq) as bf16 (one block per token)
// ---------------------------------------------------------------------------
__global__ __launch_bounds__(256)
void k_build_hcat(const int* __restrict__ tti, const u16* __restrict__ tc_bf,
                  const u16* __restrict__ rowh_bf, const float* __restrict__ seq,
                  u16* __restrict__ hcat)
{
    int bl = blockIdx.x;
    int b  = bl >> 10;
    int cid = tti[bl * 3 + 1];
    int rid = tti[bl * 3 + 2];
    u16* dst        = hcat + (size_t)bl * 2304;
    const u16* cs   = tc_bf   + (size_t)(b * 256 + cid) * 768;
    const u16* rs   = rowh_bf + (size_t)(b * 256 + rid) * 768;
    const float* ss = seq     + (size_t)bl * 768;
    for (int d = threadIdx.x; d < 768; d += 256) {
        dst[d]        = cs[d];
        dst[768 + d]  = rs[d];
        dst[1536 + d] = f2bf(ss[d]);
    }
}

// ---------------------------------------------------------------------------
// predictions = h @ W_comb + b_comb; also softmax[...,1]  (one block per token)
// ---------------------------------------------------------------------------
__global__ __launch_bounds__(256)
void k_pred_comb(const u16* __restrict__ h, const float* __restrict__ wcomb,
                 const float* __restrict__ bcomb, float* __restrict__ preds,
                 float* __restrict__ soft_out, int K)
{
    __shared__ float s0[256], s1[256];
    int row = blockIdx.x;
    float p0 = 0.0f, p1 = 0.0f;
    for (int k = threadIdx.x; k < K; k += 256) {
        float hv = bf2f(h[(size_t)row * K + k]);
        p0 += hv * wcomb[k * 2 + 0];
        p1 += hv * wcomb[k * 2 + 1];
    }
    s0[threadIdx.x] = p0; s1[threadIdx.x] = p1;
    __syncthreads();
    for (int s = 128; s > 0; s >>= 1) {
        if (threadIdx.x < s) { s0[threadIdx.x] += s0[threadIdx.x + s];
                               s1[threadIdx.x] += s1[threadIdx.x + s]; }
        __syncthreads();
    }
    if (threadIdx.x == 0) {
        float a = s0[0] + bcomb[0], b = s1[0] + bcomb[1];
        preds[row * 2 + 0] = a;
        preds[row * 2 + 1] = b;
        float m = fmaxf(a, b);
        float e0 = __expf(a - m), e1 = __expf(b - m);
        soft_out[row] = e1 / (e0 + e1);
    }
}

// ---------------------------------------------------------------------------
// prediction_row = seg_sum(predictions, row_ids)[:, :, 1]   (block per (b,c))
// ---------------------------------------------------------------------------
__global__ __launch_bounds__(256)
void k_row_segsum(const int* __restrict__ tti, const float* __restrict__ preds,
                  float* __restrict__ out)
{
    __shared__ float sm[256];
    int b = blockIdx.x >> 8;
    int c = blockIdx.x & 255;
    float acc = 0.0f;
    for (int l = threadIdx.x; l < 1024; l += 256) {
        int bl = (b << 10) + l;
        if (tti[bl * 3 + 2] == c) acc += preds[bl * 2 + 1];
    }
    sm[threadIdx.x] = acc;
    __syncthreads();
    for (int s = 128; s > 0; s >>= 1) {
        if (threadIdx.x < s) sm[threadIdx.x] += sm[threadIdx.x + s];
        __syncthreads();
    }
    if (threadIdx.x == 0) out[blockIdx.x] = sm[0];
}

// ---------------------------------------------------------------------------
// pooled = gelu(pooled_output @ W_agg + b_agg)   (one thread per output)
// ---------------------------------------------------------------------------
__global__ __launch_bounds__(256)
void k_pooled(const float* __restrict__ po, const float* __restrict__ W,
              const float* __restrict__ b, float* __restrict__ pooled)
{
    int idx = blockIdx.x * 256 + threadIdx.x;   // 16*768 outputs
    if (idx >= 16 * 768) return;
    int bb = idx / 768, j = idx % 768;
    float acc = 0.0f;
    for (int k = 0; k < 768; k++) acc += po[bb * 768 + k] * W[k * 768 + j];
    pooled[idx] = gelu_f(acc + b[j]);
}

__global__ __launch_bounds__(128)
void k_agg(const float* __restrict__ pooled, const float* __restrict__ W,
           const float* __restrict__ b, float* __restrict__ out)
{
    int idx = threadIdx.x;                       // 96 outputs
    if (idx >= 96) return;
    int bb = idx / 6, j = idx % 6;
    float acc = 0.0f;
    for (int k = 0; k < 768; k++) acc += pooled[bb * 768 + k] * W[k * 6 + j];
    out[idx] = acc + b[j];
}

// ---------------------------------------------------------------------------
// Host launcher
// ---------------------------------------------------------------------------
extern "C" void kernel_launch(void* const* d_in, const int* in_sizes, int n_in,
                              void* d_out, int out_size, void* d_ws, size_t ws_size,
                              hipStream_t stream)
{
    (void)in_sizes; (void)n_in; (void)out_size; (void)ws_size;
    const int*   input_ids = (const int*)  d_in[0];
    const int*   tti       = (const int*)  d_in[1];
    const float* pooled_in = (const float*)d_in[2];
    const float* seq       = (const float*)d_in[3];
    const float* W_agg     = (const float*)d_in[4];
    const float* b_agg     = (const float*)d_in[5];
    const float* W_cls_agg = (const float*)d_in[6];
    const float* b_cls_agg = (const float*)d_in[7];
    const float* W_col     = (const float*)d_in[8];
    const float* b_col     = (const float*)d_in[9];
    const float* W_cls_col = (const float*)d_in[10];
    const float* b_cls_col = (const float*)d_in[11];
    const float* W_tc      = (const float*)d_in[12];
    const float* b_tc      = (const float*)d_in[13];
    const float* W_row     = (const float*)d_in[14];
    const float* b_row     = (const float*)d_in[15];
    const float* W_c1      = (const float*)d_in[16];
    const float* b_c1      = (const float*)d_in[17];
    const float* W_c2      = (const float*)d_in[18];
    const float* b_c2      = (const float*)d_in[19];
    const float* W_c3      = (const float*)d_in[20];
    const float* b_c3      = (const float*)d_in[21];
    const float* W_comb    = (const float*)d_in[22];
    const float* b_comb    = (const float*)d_in[23];

    float* out = (float*)d_out;
    // output layout: pc [0,4096)  pr [4096,8192)  soft [8192,24576)  agg [24576,24672)
    float* out_pc   = out;
    float* out_pr   = out + 4096;
    float* out_soft = out + 8192;
    float* out_agg  = out + 24576;

    // --- workspace carve-up (256B aligned) ---
    size_t off = 0;
    char* base = (char*)d_ws;
    auto alloc = [&](size_t bytes) -> char* {
        char* p = base + off;
        off += (bytes + 255) & ~(size_t)255;
        return p;
    };
    const int CELL = 4096 * 768;          // (B*C, H)
    const int HH   = 768 * 768;
    const int H3   = 2304 * 2304;
    const int TOK  = 16384;               // B*L

    float* col_cell   = (float*)alloc((size_t)CELL * 4);
    float* row_cell   = (float*)alloc((size_t)CELL * 4);
    u16*   col_bf     = (u16*)  alloc((size_t)CELL * 2);
    u16*   row_bf     = (u16*)  alloc((size_t)CELL * 2);
    u16*   wcolT_bf   = (u16*)  alloc((size_t)HH * 2);   // W^T (N x K)
    u16*   wtcT_bf    = (u16*)  alloc((size_t)HH * 2);
    u16*   wrowT_bf   = (u16*)  alloc((size_t)HH * 2);
    u16*   wc1T_bf    = (u16*)  alloc((size_t)H3 * 2);
    u16*   wc2T_bf    = (u16*)  alloc((size_t)H3 * 2);
    u16*   wc3T_bf    = (u16*)  alloc((size_t)H3 * 2);
    u16*   colh_bf    = (u16*)  alloc((size_t)CELL * 2);
    u16*   tc_bf      = (u16*)  alloc((size_t)CELL * 2);
    u16*   rowh_bf    = (u16*)  alloc((size_t)CELL * 2);
    u16*   hA         = (u16*)  alloc((size_t)TOK * 2304 * 2);
    u16*   hB         = (u16*)  alloc((size_t)TOK * 2304 * 2);
    float* preds      = (float*)alloc((size_t)TOK * 2 * 4);
    float* pooled_ws  = (float*)alloc((size_t)16 * 768 * 4);

    auto cgrid = [](int n) { int g = (n + 255) / 256; return g > 4096 ? 4096 : g; };

    // --- 0) zero segment-sum accumulators ---
    k_zero_f32<<<cgrid(CELL), 256, 0, stream>>>(col_cell, CELL);
    k_zero_f32<<<cgrid(CELL), 256, 0, stream>>>(row_cell, CELL);

    // --- 1) transpose-convert weights to bf16 (W^T, N x K) ---
    dim3 tpH(768 / 32, 768 / 32), tp3(2304 / 32, 2304 / 32);
    k_f32_to_bf16_T<<<tpH, 256, 0, stream>>>(W_col, wcolT_bf, 768, 768);
    k_f32_to_bf16_T<<<tpH, 256, 0, stream>>>(W_tc,  wtcT_bf,  768, 768);
    k_f32_to_bf16_T<<<tpH, 256, 0, stream>>>(W_row, wrowT_bf, 768, 768);
    k_f32_to_bf16_T<<<tp3, 256, 0, stream>>>(W_c1,  wc1T_bf,  2304, 2304);
    k_f32_to_bf16_T<<<tp3, 256, 0, stream>>>(W_c2,  wc2T_bf,  2304, 2304);
    k_f32_to_bf16_T<<<tp3, 256, 0, stream>>>(W_c3,  wc3T_bf,  2304, 2304);

    // --- 2) pooled/agg path (tiny, independent) ---
    k_pooled<<<(16 * 768 + 255) / 256, 256, 0, stream>>>(pooled_in, W_agg, b_agg, pooled_ws);
    k_agg<<<1, 128, 0, stream>>>(pooled_ws, W_cls_agg, b_cls_agg, out_agg);

    // --- 3) masked segment sums into cells ---
    k_scatter_cells<<<TOK, 256, 0, stream>>>(input_ids, tti, seq, col_cell, row_cell);
    k_f32_to_bf16<<<cgrid(CELL), 256, 0, stream>>>(col_cell, col_bf, CELL);
    k_f32_to_bf16<<<cgrid(CELL), 256, 0, stream>>>(row_cell, row_bf, CELL);

    // --- 4) cell GEMMs (M=4096, N=768, K=768), gelu fused ---
    dim3 g768(768 / BN, 4096 / BM);
    k_gemm_bf16<<<g768, 256, 0, stream>>>(col_bf, wcolT_bf, b_col, colh_bf, 4096, 768, 768, 1);
    k_gemm_bf16<<<g768, 256, 0, stream>>>(col_bf, wtcT_bf,  b_tc,  tc_bf,   4096, 768, 768, 1);
    k_gemm_bf16<<<g768, 256, 0, stream>>>(row_bf, wrowT_bf, b_row, rowh_bf, 4096, 768, 768, 1);

    // --- 5) prediction_column ---
    k_rowdot<<<4096, 256, 0, stream>>>(colh_bf, W_cls_col, b_cls_col, out_pc, 768);

    // --- 6) gather + concat -> h (bf16) ---
    k_build_hcat<<<TOK, 256, 0, stream>>>(tti, tc_bf, rowh_bf, seq, hA);

    // --- 7) big GEMM stack (M=16384, N=K=2304), gelu fused ---
    dim3 g2304(2304 / BN, 16384 / BM);
    k_gemm_bf16<<<g2304, 256, 0, stream>>>(hA, wc1T_bf, b_c1, hB, 16384, 2304, 2304, 1);
    k_gemm_bf16<<<g2304, 256, 0, stream>>>(hB, wc2T_bf, b_c2, hA, 16384, 2304, 2304, 1);
    k_gemm_bf16<<<g2304, 256, 0, stream>>>(hA, wc3T_bf, b_c3, hB, 16384, 2304, 2304, 1);

    // --- 8) final projection + softmax ---
    k_pred_comb<<<TOK, 256, 0, stream>>>(hB, W_comb, b_comb, preds, out_soft, 2304);

    // --- 9) row segment sum of predictions[...,1] ---
    k_row_segsum<<<4096, 256, 0, stream>>>(tti, preds, out_pr);
}
